// single_head_split_window_attention_1d_48387101557135
// MI455X (gfx1250) — matrix operand order repaired and checked by reading, not compiled, not run on hardware
//
#include <hip/hip_runtime.h>

// ---- CDNA5 WMMA vector types ----
typedef __attribute__((ext_vector_type(16))) __bf16 v16bf;
typedef __attribute__((ext_vector_type(8)))  __bf16 v8bf;
typedef __attribute__((ext_vector_type(8)))  float  v8f;
typedef __attribute__((ext_vector_type(4)))  float  v4f;

// Problem constants (reference: b=4, h=128, w=512, c=128, num_splits=8)
constexpr int BH     = 512;   // b*h
constexpr int W      = 512;
constexpr int C      = 128;
constexpr int WIN    = 64;
constexpr int NSPLIT = 8;
constexpr int SHIFT  = 32;    // win/2
constexpr int VT_STR = 72;    // padded stride for transposed V (16B-aligned, conflict-reducing)

__device__ __forceinline__ v16bf cat8(v8bf lo, v8bf hi) {
  return __builtin_shufflevector(lo, hi, 0,1,2,3,4,5,6,7,8,9,10,11,12,13,14,15);
}

__global__ __launch_bounds__(128)
void swin_attn_wmma_kernel(const float* __restrict__ q,
                           const float* __restrict__ k,
                           const float* __restrict__ v,
                           const float* __restrict__ mask,
                           float* __restrict__ out) {
  __shared__ __bf16 Qs[WIN][C];        // 16 KB
  __shared__ __bf16 Ks[WIN][C];        // 16 KB
  __shared__ __bf16 Vt[C][VT_STR];     // 18 KB (V transposed: Vt[channel][k])
  __shared__ float  Ms[WIN][WIN];      // 16 KB (mask window)
  __shared__ __bf16 Ps[WIN][WIN];      //  8 KB (softmax probabilities)

  const int tid  = threadIdx.x;
  const int wave = tid >> 5;
  const int lane = tid & 31;
  const int bid  = blockIdx.x;
  const int s    = bid & (NSPLIT - 1);
  const int bh   = bid >> 3;

  // ------------- load phase: shifted Q/K/V window + mask -> LDS (bf16) -------------
  // q/k/v rows are touched exactly once device-wide -> non-temporal loads (TH=NT),
  // keep L2 free for the mask (reused by all 4096 workgroups).
  // 64 rows x 32 float4 per matrix; 128 threads -> 16 iterations
  #pragma unroll 4
  for (int it = 0; it < 16; ++it) {
    int idx = it * 128 + tid;
    int r   = idx >> 5;          // window row 0..63
    int c4  = idx & 31;          // float4 index within row
    int wr  = (s * WIN + r + SHIFT) & (W - 1);       // roll(-shift)
    size_t goff = ((size_t)bh * W + wr) * C + (size_t)c4 * 4;
    v4f fq = __builtin_nontemporal_load(reinterpret_cast<const v4f*>(q + goff));
    v4f fk = __builtin_nontemporal_load(reinterpret_cast<const v4f*>(k + goff));
    v4f fv = __builtin_nontemporal_load(reinterpret_cast<const v4f*>(v + goff));
    int c = c4 * 4;
    Qs[r][c+0] = (__bf16)fq.x; Qs[r][c+1] = (__bf16)fq.y;
    Qs[r][c+2] = (__bf16)fq.z; Qs[r][c+3] = (__bf16)fq.w;
    Ks[r][c+0] = (__bf16)fk.x; Ks[r][c+1] = (__bf16)fk.y;
    Ks[r][c+2] = (__bf16)fk.z; Ks[r][c+3] = (__bf16)fk.w;
    Vt[c+0][r] = (__bf16)fv.x; Vt[c+1][r] = (__bf16)fv.y;
    Vt[c+2][r] = (__bf16)fv.z; Vt[c+3][r] = (__bf16)fv.w;
  }
  // mask window: 4096 floats = 1024 float4; 8 iterations (regular-temporal: L2-resident)
  #pragma unroll
  for (int it = 0; it < 8; ++it) {
    int idx = it * 128 + tid;
    v4f fm = *reinterpret_cast<const v4f*>(mask + (size_t)s * (WIN * WIN) + (size_t)idx * 4);
    *reinterpret_cast<v4f*>(&Ms[0][0] + idx * 4) = fm;
  }
  __syncthreads();

  // ------------- fragment coordinates (ISA §7.12.2 layouts) -------------
  const int mrow = wave * 16 + (lane & 15);   // A-fragment: lane -> matrix row
  const int kbA  = (lane >> 4) * 8;           // A: lane-half K split (0..7 | 8..15, +16 for hi elems)
  const int ncol = (lane & 15);               // B/D: lane -> column within 16-wide tile
  const int kbB  = (lane >> 4) * 8;           // B: lane-half K split (0..7 | 8..15, +16 for hi elems)

  // ------------- S = Q K^T : 4 N-tiles x 4 K-chunks of 32 -------------
  v8f acc[4] = {};
  #pragma unroll
  for (int c0 = 0; c0 < C; c0 += 32) {
    v8bf alo = *reinterpret_cast<const v8bf*>(&Qs[mrow][c0 + kbA]);
    v8bf ahi = *reinterpret_cast<const v8bf*>(&Qs[mrow][c0 + 16 + kbA]);
    v16bf a = cat8(alo, ahi);
    #pragma unroll
    for (int t = 0; t < 4; ++t) {
      const __bf16* bp = &Ks[t * 16 + ncol][c0 + kbB];   // B(k,n) = K[n][c0+k]
      v16bf b = cat8(*reinterpret_cast<const v8bf*>(bp),
                     *reinterpret_cast<const v8bf*>(bp + 16));
      acc[t] = __builtin_amdgcn_wmma_f32_16x16x32_bf16(
          false, a, false, b, (short)0, acc[t], false, false);
    }
  }

  // ------------- softmax over K (rows live in lane-halves of D fragments) -------------
  const float scale = 0.0883883476483184f;  // 1/sqrt(128)
  const int rbase = wave * 16 + ((lane >> 4) ? 8 : 0);
  #pragma unroll
  for (int r = 0; r < 8; ++r) {
    int row = rbase + r;
    float vals[4];
    #pragma unroll
    for (int t = 0; t < 4; ++t)
      vals[t] = acc[t][r] * scale + Ms[row][t * 16 + ncol];
    float m = fmaxf(fmaxf(vals[0], vals[1]), fmaxf(vals[2], vals[3]));
    m = fmaxf(m, __shfl_xor(m, 1, 16));
    m = fmaxf(m, __shfl_xor(m, 2, 16));
    m = fmaxf(m, __shfl_xor(m, 4, 16));
    m = fmaxf(m, __shfl_xor(m, 8, 16));
    float p[4], sum = 0.f;
    #pragma unroll
    for (int t = 0; t < 4; ++t) { p[t] = __expf(vals[t] - m); sum += p[t]; }
    sum += __shfl_xor(sum, 1, 16);
    sum += __shfl_xor(sum, 2, 16);
    sum += __shfl_xor(sum, 4, 16);
    sum += __shfl_xor(sum, 8, 16);
    float inv = 1.0f / sum;
    #pragma unroll
    for (int t = 0; t < 4; ++t)
      Ps[row][t * 16 + ncol] = (__bf16)(p[t] * inv);
  }
  __syncthreads();   // cross-lane LDS visibility before fragment reloads

  // ------------- O = P V : 8 channel-tiles x 2 K-chunks of 32 -------------
  v8f o[8] = {};
  #pragma unroll
  for (int k0 = 0; k0 < WIN; k0 += 32) {
    v8bf plo = *reinterpret_cast<const v8bf*>(&Ps[mrow][k0 + kbA]);
    v8bf phi = *reinterpret_cast<const v8bf*>(&Ps[mrow][k0 + 16 + kbA]);
    v16bf a = cat8(plo, phi);
    #pragma unroll
    for (int t = 0; t < 8; ++t) {
      const __bf16* bp = &Vt[t * 16 + ncol][k0 + kbB];   // B(k,n) = V[k][n] = Vt[n][k]
      v16bf b = cat8(*reinterpret_cast<const v8bf*>(bp),
                     *reinterpret_cast<const v8bf*>(bp + 16));
      o[t] = __builtin_amdgcn_wmma_f32_16x16x32_bf16(
          false, a, false, b, (short)0, o[t], false, false);
    }
  }

  // ------------- store (roll(+shift) == same index map as load); write-once -> NT -------------
  #pragma unroll
  for (int r = 0; r < 8; ++r) {
    int row = rbase + r;
    int wr  = (s * WIN + row + SHIFT) & (W - 1);
    float* op = out + ((size_t)bh * W + wr) * C;
    #pragma unroll
    for (int t = 0; t < 8; ++t)
      __builtin_nontemporal_store(o[t][r], op + t * 16 + ncol);
  }
}

extern "C" void kernel_launch(void* const* d_in, const int* in_sizes, int n_in,
                              void* d_out, int out_size, void* d_ws, size_t ws_size,
                              hipStream_t stream) {
  (void)in_sizes; (void)n_in; (void)out_size; (void)d_ws; (void)ws_size;
  const float* q    = (const float*)d_in[0];
  const float* k    = (const float*)d_in[1];
  const float* v    = (const float*)d_in[2];
  const float* mask = (const float*)d_in[3];
  float* out = (float*)d_out;
  // 512 (b*h) rows x 8 windows = 4096 workgroups, 4 waves each
  swin_attn_wmma_kernel<<<dim3(BH * NSPLIT), dim3(128), 0, stream>>>(q, k, v, mask, out);
}
